// GraphMathLayer_42099269435541
// MI455X (gfx1250) — matrix operand order repaired and checked by reading, not compiled, not run on hardware
//
#include <hip/hip_runtime.h>

#define HID 128

typedef __attribute__((ext_vector_type(16))) __bf16 v16bf;
typedef __attribute__((ext_vector_type(8)))  float  v8f;
typedef __attribute__((ext_vector_type(4)))  int    vi4;

union FragBF {
    v16bf v;
    vi4   q[2];
};

__device__ __forceinline__ unsigned short f2bf(float f) {
    unsigned u = __float_as_uint(f);
    u += 0x7FFFu + ((u >> 16) & 1u);   // round-to-nearest-even
    return (unsigned short)(u >> 16);
}

__device__ __forceinline__ v8f v8f_splat(float f) {
    v8f r;
    #pragma unroll
    for (int i = 0; i < 8; ++i) r[i] = f;
    return r;
}

// ---------------------------------------------------------------------------
// Zero the scatter-accumulator (float4-vectorized).
__global__ void zero4_kernel(float4* __restrict__ p, int n4) {
    int i = blockIdx.x * blockDim.x + threadIdx.x;
    if (i < n4) p[i] = make_float4(0.f, 0.f, 0.f, 0.f);
}

// ---------------------------------------------------------------------------
// Pack a row-major f32 weight matrix W[K x 128] into per-lane bf16 B-fragment
// order for v_wmma_f32_16x16x32_bf16.
// Packed dword index t = ((kb*8 + nt)*32 + lane)*8 + j, where lane = khi*16+n,
// holds bf16(W[k0][col]) | bf16(W[k0+1][col]) << 16 with
// k0 = kb*32 + (j>>2)*16 + khi*8 + 2*(j&3), col = nt*16 + n.
__global__ void pack_weights_kernel(const float* __restrict__ W,
                                    unsigned* __restrict__ out, int kblocks) {
    int t = blockIdx.x * blockDim.x + threadIdx.x;
    int total = kblocks * 8 * 32 * 8;
    if (t >= total) return;
    int j    =  t        & 7;
    int lane = (t >> 3)  & 31;
    int nt   = (t >> 8)  & 7;
    int kb   =  t >> 11;
    int khi = lane >> 4;
    int n   = lane & 15;
    int k0  = kb * 32 + (j >> 2) * 16 + khi * 8 + 2 * (j & 3);
    int col = nt * 16 + n;
    unsigned lo = f2bf(W[(size_t)k0 * HID + col]);
    unsigned hi = f2bf(W[(size_t)(k0 + 1) * HID + col]);
    out[t] = lo | (hi << 16);
}

// ---------------------------------------------------------------------------
// Edge kernel: message MLP over 16 edges per wave + f32 atomic scatter-add.
__global__ __launch_bounds__(256) void edge_kernel(
    const float* __restrict__ x, const long long* __restrict__ ei,
    const float* __restrict__ ea,
    const unsigned* __restrict__ w1p, const float* __restrict__ b1,
    const unsigned* __restrict__ w2p, const float* __restrict__ b2,
    float* __restrict__ agg, long long n_edges) {
    // per-wave 5 KB: panel (16x32 bf16) at [0,512), H (16x128 bf16) at [512,2560)
    __shared__ __align__(16) unsigned short lds[8][2560];

    const int tid  = threadIdx.x;
    const int wave = tid >> 5;
    const int lane = tid & 31;
    const int n    = lane & 15;
    const int hi   = lane >> 4;

    const long long e0   = (long long)blockIdx.x * 128 + wave * 16;
    const long long* srcp = ei;
    const long long* dstp = ei + n_edges;

    unsigned short* panel = &lds[wave][0];
    unsigned short* H     = &lds[wave][512];

    // GEMM1: [16x384] x [384x128], accumulators seeded with bias b1
    v8f acc[8];
    #pragma unroll
    for (int nt = 0; nt < 8; ++nt) acc[nt] = v8f_splat(b1[nt * 16 + n]);

    for (int kb = 0; kb < 12; ++kb) {
        const int part  = kb >> 2;           // 0: x[src], 1: x[dst], 2: edge_attr
        const int cbase = (kb & 3) * 32;
        // gather 16 rows x 32 cols (f32 -> bf16), 4 float4 chunks per lane
        #pragma unroll
        for (int i = 0; i < 4; ++i) {
            int c   = lane + 32 * i;         // chunk 0..127
            int row = c >> 3;
            int q   = c & 7;
            long long e = e0 + row;
            if (e >= n_edges) e = 0;
            const float* srow;
            if (part == 0)      srow = x  + (size_t)srcp[e] * HID;
            else if (part == 1) srow = x  + (size_t)dstp[e] * HID;
            else                srow = ea + (size_t)e * HID;
            float4 v = *(const float4*)(srow + cbase + q * 4);
            unsigned short* d = panel + row * 32 + q * 4;
            d[0] = f2bf(v.x); d[1] = f2bf(v.y); d[2] = f2bf(v.z); d[3] = f2bf(v.w);
        }
        __syncthreads();

        FragBF a;
        a.q[0] = *(const vi4*)(panel + n * 32 + 8 * hi);
        a.q[1] = *(const vi4*)(panel + n * 32 + 16 + 8 * hi);

        #pragma unroll
        for (int nt = 0; nt < 8; ++nt) {
            FragBF b;
            const vi4* bp = (const vi4*)w1p + (size_t)((kb * 8 + nt) * 32 + lane) * 2;
            b.q[0] = bp[0];
            b.q[1] = bp[1];
            acc[nt] = __builtin_amdgcn_wmma_f32_16x16x32_bf16(
                false, a.v, false, b.v, (short)0, acc[nt], false, false);
        }
        __syncthreads();
    }

    // ReLU -> bf16 H in LDS
    #pragma unroll
    for (int nt = 0; nt < 8; ++nt)
        #pragma unroll
        for (int j = 0; j < 8; ++j) {
            float v = acc[nt][j];
            v = v > 0.f ? v : 0.f;
            H[(j + 8 * hi) * HID + nt * 16 + n] = f2bf(v);
        }
    __syncthreads();

    // GEMM2: [16x128] x [128x128], accumulators seeded with b2
    v8f acc2[8];
    #pragma unroll
    for (int nt = 0; nt < 8; ++nt) acc2[nt] = v8f_splat(b2[nt * 16 + n]);

    for (int kb = 0; kb < 4; ++kb) {
        FragBF a;
        a.q[0] = *(const vi4*)(H + n * HID + kb * 32 + 8 * hi);
        a.q[1] = *(const vi4*)(H + n * HID + kb * 32 + 16 + 8 * hi);
        #pragma unroll
        for (int nt = 0; nt < 8; ++nt) {
            FragBF b;
            const vi4* bp = (const vi4*)w2p + (size_t)((kb * 8 + nt) * 32 + lane) * 2;
            b.q[0] = bp[0];
            b.q[1] = bp[1];
            acc2[nt] = __builtin_amdgcn_wmma_f32_16x16x32_bf16(
                false, a.v, false, b.v, (short)0, acc2[nt], false, false);
        }
    }
    __syncthreads();

    // scatter-add messages into aggregated[dst]
    int dsti[8];
    bool valid[8];
    #pragma unroll
    for (int j = 0; j < 8; ++j) {
        long long e = e0 + j + 8 * hi;
        valid[j] = (e < n_edges);
        dsti[j]  = valid[j] ? (int)dstp[e] : 0;
    }
    #pragma unroll
    for (int nt = 0; nt < 8; ++nt)
        #pragma unroll
        for (int j = 0; j < 8; ++j)
            if (valid[j])
                unsafeAtomicAdd(&agg[(size_t)dsti[j] * HID + nt * 16 + n], acc2[nt][j]);
}

// ---------------------------------------------------------------------------
// Node kernel: update MLP + residual + LayerNorm, 16 nodes per wave.
__global__ __launch_bounds__(256) void node_kernel(
    const float* __restrict__ x, const float* __restrict__ agg,
    const unsigned* __restrict__ u1p, const float* __restrict__ b1,
    const unsigned* __restrict__ u2p, const float* __restrict__ b2,
    const float* __restrict__ gamma, const float* __restrict__ beta,
    float* __restrict__ out, int n_nodes) {
    // per-wave 8 KB union: panel [0,512), H [512,2560) (bf16); R = f32[16*128] over [0,4096)
    __shared__ __align__(16) unsigned short lds[8][4096];

    const int tid  = threadIdx.x;
    const int wave = tid >> 5;
    const int lane = tid & 31;
    const int n    = lane & 15;
    const int hi   = lane >> 4;

    const int node0 = blockIdx.x * 128 + wave * 16;

    unsigned short* panel = &lds[wave][0];
    unsigned short* H     = &lds[wave][512];
    float*          R     = (float*)&lds[wave][0];

    // GEMM1: [16x256]([x|agg]) x [256x128]
    v8f acc[8];
    #pragma unroll
    for (int nt = 0; nt < 8; ++nt) acc[nt] = v8f_splat(b1[nt * 16 + n]);

    for (int kb = 0; kb < 8; ++kb) {
        const int part  = kb >> 2;          // 0: x, 1: aggregated
        const int cbase = (kb & 3) * 32;
        #pragma unroll
        for (int i = 0; i < 4; ++i) {
            int c   = lane + 32 * i;
            int row = c >> 3;
            int q   = c & 7;
            int nid = node0 + row;
            if (nid >= n_nodes) nid = 0;
            const float* srow = (part == 0 ? x : agg) + (size_t)nid * HID;
            float4 v = *(const float4*)(srow + cbase + q * 4);
            unsigned short* d = panel + row * 32 + q * 4;
            d[0] = f2bf(v.x); d[1] = f2bf(v.y); d[2] = f2bf(v.z); d[3] = f2bf(v.w);
        }
        __syncthreads();

        FragBF a;
        a.q[0] = *(const vi4*)(panel + n * 32 + 8 * hi);
        a.q[1] = *(const vi4*)(panel + n * 32 + 16 + 8 * hi);

        #pragma unroll
        for (int nt = 0; nt < 8; ++nt) {
            FragBF b;
            const vi4* bp = (const vi4*)u1p + (size_t)((kb * 8 + nt) * 32 + lane) * 2;
            b.q[0] = bp[0];
            b.q[1] = bp[1];
            acc[nt] = __builtin_amdgcn_wmma_f32_16x16x32_bf16(
                false, a.v, false, b.v, (short)0, acc[nt], false, false);
        }
        __syncthreads();
    }

    #pragma unroll
    for (int nt = 0; nt < 8; ++nt)
        #pragma unroll
        for (int j = 0; j < 8; ++j) {
            float v = acc[nt][j];
            v = v > 0.f ? v : 0.f;
            H[(j + 8 * hi) * HID + nt * 16 + n] = f2bf(v);
        }
    __syncthreads();

    // GEMM2: [16x128] x [128x128]
    v8f acc2[8];
    #pragma unroll
    for (int nt = 0; nt < 8; ++nt) acc2[nt] = v8f_splat(b2[nt * 16 + n]);

    for (int kb = 0; kb < 4; ++kb) {
        FragBF a;
        a.q[0] = *(const vi4*)(H + n * HID + kb * 32 + 8 * hi);
        a.q[1] = *(const vi4*)(H + n * HID + kb * 32 + 16 + 8 * hi);
        #pragma unroll
        for (int nt = 0; nt < 8; ++nt) {
            FragBF b;
            const vi4* bp = (const vi4*)u2p + (size_t)((kb * 8 + nt) * 32 + lane) * 2;
            b.q[0] = bp[0];
            b.q[1] = bp[1];
            acc2[nt] = __builtin_amdgcn_wmma_f32_16x16x32_bf16(
                false, a.v, false, b.v, (short)0, acc2[nt], false, false);
        }
    }
    __syncthreads();   // protect H region before R overwrites it

    // residual add, stage f32 rows into R
    #pragma unroll
    for (int j = 0; j < 8; ++j) {
        int r   = j + 8 * hi;
        int nid = node0 + r;
        if (nid >= n_nodes) nid = 0;
        #pragma unroll
        for (int nt = 0; nt < 8; ++nt) {
            int col = nt * 16 + n;
            R[r * HID + col] = acc2[nt][j] + x[(size_t)nid * HID + col];
        }
    }
    __syncthreads();

    // LayerNorm row stats: lanes 0..15 each own one row, broadcast via shfl
    float mu = 0.f, rs = 0.f;
    if (lane < 16) {
        float s = 0.f, s2 = 0.f;
        const float* row = R + lane * HID;
        for (int c = 0; c < HID; ++c) {
            float v = row[c];
            s += v;
            s2 += v * v;
        }
        mu = s * (1.f / HID);
        float var = s2 * (1.f / HID) - mu * mu;
        rs = rsqrtf(var + 1e-5f);
    }

    for (int r = 0; r < 16; ++r) {
        int nid = node0 + r;
        if (nid >= n_nodes) continue;
        float m = __shfl(mu, r);
        float s = __shfl(rs, r);
        for (int c = lane; c < HID; c += 32) {
            float v = (R[r * HID + c] - m) * s * gamma[c] + beta[c];
            out[(size_t)nid * HID + c] = v;
        }
    }
}

// ---------------------------------------------------------------------------
extern "C" void kernel_launch(void* const* d_in, const int* in_sizes, int n_in,
                              void* d_out, int out_size, void* d_ws, size_t ws_size,
                              hipStream_t stream) {
    const float*     x     = (const float*)d_in[0];
    const long long* ei    = (const long long*)d_in[1];
    const float*     ea    = (const float*)d_in[2];
    const float*     msgW1 = (const float*)d_in[3];
    const float*     msgb1 = (const float*)d_in[4];
    const float*     msgW2 = (const float*)d_in[5];
    const float*     msgb2 = (const float*)d_in[6];
    const float*     updW1 = (const float*)d_in[7];
    const float*     updb1 = (const float*)d_in[8];
    const float*     updW2 = (const float*)d_in[9];
    const float*     updb2 = (const float*)d_in[10];
    const float*     g     = (const float*)d_in[11];
    const float*     bta   = (const float*)d_in[12];

    const int       n_nodes = in_sizes[0] / HID;
    const long long n_edges = (long long)in_sizes[2] / HID;

    // workspace layout: aggregated[N*128] f32, then 4 packed weight blocks
    char*  ws  = (char*)d_ws;
    float* agg = (float*)ws;
    size_t off = ((size_t)n_nodes * HID * sizeof(float) + 255) & ~(size_t)255;
    unsigned* w1p = (unsigned*)(ws + off); off += (size_t)12 * 2048 * 4;
    unsigned* w2p = (unsigned*)(ws + off); off += (size_t)4  * 2048 * 4;
    unsigned* u1p = (unsigned*)(ws + off); off += (size_t)8  * 2048 * 4;
    unsigned* u2p = (unsigned*)(ws + off); off += (size_t)4  * 2048 * 4;

    const int n4 = (n_nodes * HID) / 4;
    zero4_kernel<<<(n4 + 255) / 256, 256, 0, stream>>>((float4*)agg, n4);

    pack_weights_kernel<<<(12 * 2048 + 255) / 256, 256, 0, stream>>>(msgW1, w1p, 12);
    pack_weights_kernel<<<(4  * 2048 + 255) / 256, 256, 0, stream>>>(msgW2, w2p, 4);
    pack_weights_kernel<<<(8  * 2048 + 255) / 256, 256, 0, stream>>>(updW1, u1p, 8);
    pack_weights_kernel<<<(4  * 2048 + 255) / 256, 256, 0, stream>>>(updW2, u2p, 4);

    const int eblocks = (int)((n_edges + 127) / 128);
    edge_kernel<<<eblocks, 256, 0, stream>>>(x, ei, ea, w1p, msgb1, w2p, msgb2,
                                             agg, n_edges);

    const int nblocks = (n_nodes + 127) / 128;
    node_kernel<<<nblocks, 256, 0, stream>>>(x, agg, u1p, updb1, u2p, updb2,
                                             g, bta, (float*)d_out, n_nodes);
}
